// RNA_Encoder_32865089749703
// MI455X (gfx1250) — matrix operand order repaired and checked by reading, not compiled, not run on hardware
//
#include <hip/hip_runtime.h>
#include <hip/hip_bf16.h>

// ---------------------------------------------------------------------------
// RNA Mamba encoder for MI455X (gfx1250, wave32).
//  - All dense GEMMs -> v_wmma_f32_16x16x32_f16, f32 accumulate,
//    2x2 16x16 tiles per wave; block A-tile (32 x K<=512) staged in LDS
//    via global_load_async_to_lds_b128 (ASYNCcnt) when available.
//  - Token GEMM (M=64,N=4000,K=4000) split-K=8 with deterministic f32 reduce.
//  - f16 activations/weights in workspace (halve HBM bytes @ 23.3 TB/s).
//  - Selective scan: state dim (16) spread across 16 lanes, 1 exp/lane/step,
//    shfl_xor reduction, delta-projection + softplus + SiLU gate fused in.
// ---------------------------------------------------------------------------

typedef _Float16 h16;
typedef __attribute__((ext_vector_type(16))) _Float16 v16h;
typedef __attribute__((ext_vector_type(8)))  float    v8f;
typedef __attribute__((ext_vector_type(4)))  int      v4i;

#define EMBD   128
#define COMP   2000
#define TOKD   4000
#define LAYERS 4
#define DI     256
#define DSTATE 16
#define DCONV  4
#define DTRANK 8
#define BATCH  64
#define LSEQ   (COMP + 1)          // 2001
#define MROWS  (BATCH * LSEQ)      // 128064 (multiple of 32)
#define NPAD   48                  // W_x rows padded 40 -> 48
#define KSPLIT_TOK 512             // split-K chunk for token GEMM
#define NSPLIT_TOK 8               // ceil(4000/512)

#define KC_MAX     512             // max K staged in LDS per kernel
#define AROWSTRIDE (KC_MAX + 8)    // +8 halves: de-bank-conflict row padding

#if defined(__has_builtin)
#if __has_builtin(__builtin_amdgcn_global_load_async_to_lds_b128) && \
    __has_builtin(__builtin_amdgcn_s_wait_asynccnt)
#define ASYNC_LDS 1
#endif
#endif

__device__ __forceinline__ float silu_f(float x) {
    return x / (1.0f + __expf(-x));
}

// ---------------------------------------------------------------------------
// NT GEMM: C[m,n] = sum_k A[m,k]*B[n,k] (+bias[n]); f16 in, f32 WMMA accum.
// Wave computes a 32x32 tile (2x2 WMMA tiles). Block = 4 waves = 32x128 tile.
// The block's 32-row A-tile for its K-chunk is staged in LDS once (async copy
// when available), so A fragments come from LDS (4x less L2 traffic) and only
// B streams from global. blockIdx.z selects a K-split chunk; if Cpart != null
// the f32 partial goes to Cpart[z*M*N + m*N + n] (deterministic split-K).
// Requires: M%32==0, N%16==0, K%32==0, lda/ldb%8==0, (kend-kstart)<=KC_MAX.
// ---------------------------------------------------------------------------
__global__ __launch_bounds__(128) void gemm_nt_wmma(
    const h16* __restrict__ A, const h16* __restrict__ Bw,
    const float* __restrict__ bias, h16* __restrict__ C,
    float* __restrict__ Cpart,
    int M, int N, int K, int lda, int ldb, int ldc, int ksplit)
{
    __shared__ h16 shA[32 * AROWSTRIDE];

    const int lane   = threadIdx.x & 31;
    const int wave   = threadIdx.x >> 5;
    const int n0tile = blockIdx.x * 8 + wave * 2;   // first of two 16-col tiles
    const int mtile  = blockIdx.y * 2;              // two 16-row tiles
    const int m0     = mtile * 16;

    const int kstart = blockIdx.z * ksplit;
    const int kend   = (kstart + ksplit < K) ? kstart + ksplit : K;
    const int Kc     = kend - kstart;               // multiple of 32, <= KC_MAX

    // ---- cooperative A-tile stage: 32 rows x Kc halves, all waves help ----
    {
        const int vrow = Kc >> 3;                   // uint4 vectors per row
        for (int idx = threadIdx.x; idx < 32 * vrow; idx += 128) {
            int row = idx / vrow;
            int kv  = idx - row * vrow;
            const h16* src = A + (long)(m0 + row) * lda + kstart + kv * 8;
            h16*       dst = &shA[row * AROWSTRIDE + kv * 8];
#ifdef ASYNC_LDS
            __builtin_amdgcn_global_load_async_to_lds_b128(
                (__attribute__((address_space(1))) v4i*)src,
                (__attribute__((address_space(3))) v4i*)dst, 0, 0);
#else
            *reinterpret_cast<uint4*>(dst) = *reinterpret_cast<const uint4*>(src);
#endif
        }
#ifdef ASYNC_LDS
        __builtin_amdgcn_s_wait_asynccnt(0);
#endif
        __syncthreads();
    }
    if (n0tile * 16 >= N) return;                   // after the only barrier
    const bool n1ok = (n0tile + 1) * 16 < N;

    const int kh  = lane >> 4;   // 0/1 half of the wave
    const int l16 = lane & 15;

    const h16* __restrict__ brow0 = Bw + (long)(n0tile * 16 + l16) * ldb;
    const h16* __restrict__ brow1 = n1ok ? brow0 + (long)16 * ldb : brow0;
    const h16* sa0 = &shA[l16 * AROWSTRIDE + kh * 8];         // m-tile 0 rows
    const h16* sa1 = &shA[(16 + l16) * AROWSTRIDE + kh * 8];  // m-tile 1 rows

    v8f acc00 = {}, acc01 = {}, acc10 = {}, acc11 = {};
    union Pack8 { uint4 u; h16 h[8]; };

    for (int kk = 0; kk < Kc; kk += 32) {
        // A frags from LDS: rows l16/16+l16, K = kh*8..+7 and 16+kh*8..+7
        Pack8 a0l, a0h, a1l, a1h, b0l, b0h, b1l, b1h;
        a0l.u = *reinterpret_cast<const uint4*>(sa0 + kk);
        a0h.u = *reinterpret_cast<const uint4*>(sa0 + kk + 16);
        a1l.u = *reinterpret_cast<const uint4*>(sa1 + kk);
        a1h.u = *reinterpret_cast<const uint4*>(sa1 + kk + 16);
        // B frags from global (L2-resident weights): cols l16, K = kh*8..+7 and 16+kh*8..+7
        const h16* b0p = brow0 + kstart + kk + kh * 8;
        const h16* b1p = brow1 + kstart + kk + kh * 8;
        b0l.u = *reinterpret_cast<const uint4*>(b0p);
        b0h.u = *reinterpret_cast<const uint4*>(b0p + 16);
        b1l.u = *reinterpret_cast<const uint4*>(b1p);
        b1h.u = *reinterpret_cast<const uint4*>(b1p + 16);
        if (kk + 64 < Kc) {  // stream-ahead prefetch of the B weight rows
            __builtin_prefetch((const void*)(brow0 + kstart + kk + 64), 0, 3);
            __builtin_prefetch((const void*)(brow1 + kstart + kk + 64), 0, 3);
        }
        v16h a0, a1, b0, b1;
#pragma unroll
        for (int j = 0; j < 8; ++j) {
            a0[j] = a0l.h[j]; a0[j + 8] = a0h.h[j];
            a1[j] = a1l.h[j]; a1[j + 8] = a1h.h[j];
            b0[j] = b0l.h[j]; b0[j + 8] = b0h.h[j];
            b1[j] = b1l.h[j]; b1[j + 8] = b1h.h[j];
        }
        acc00 = __builtin_amdgcn_wmma_f32_16x16x32_f16(false, a0, false, b0, (short)0, acc00, false, false);
        acc01 = __builtin_amdgcn_wmma_f32_16x16x32_f16(false, a0, false, b1, (short)0, acc01, false, false);
        acc10 = __builtin_amdgcn_wmma_f32_16x16x32_f16(false, a1, false, b0, (short)0, acc10, false, false);
        acc11 = __builtin_amdgcn_wmma_f32_16x16x32_f16(false, a1, false, b1, (short)0, acc11, false, false);
    }

    // C/D layout: lane -> col l16, VGPR r -> row r + 8*kh
    const int n0 = n0tile * 16 + l16;
    const int n1 = n0 + 16;
    if (Cpart) {
        float* base = Cpart + (long)blockIdx.z * M * N;
#pragma unroll
        for (int r = 0; r < 8; ++r) {
            int mm0 = m0 + kh * 8 + r;
            int mm1 = mm0 + 16;
            base[(long)mm0 * N + n0] = acc00[r];
            base[(long)mm1 * N + n0] = acc10[r];
            if (n1ok) {
                base[(long)mm0 * N + n1] = acc01[r];
                base[(long)mm1 * N + n1] = acc11[r];
            }
        }
    } else {
        const float bv0 = bias ? bias[n0] : 0.0f;
        const float bv1 = (bias && n1ok) ? bias[n1] : 0.0f;
#pragma unroll
        for (int r = 0; r < 8; ++r) {
            int mm0 = m0 + kh * 8 + r;
            int mm1 = mm0 + 16;
            C[(long)mm0 * ldc + n0] = (h16)(acc00[r] + bv0);
            C[(long)mm1 * ldc + n0] = (h16)(acc10[r] + bv0);
            if (n1ok) {
                C[(long)mm0 * ldc + n1] = (h16)(acc01[r] + bv1);
                C[(long)mm1 * ldc + n1] = (h16)(acc11[r] + bv1);
            }
        }
    }
}

// Deterministic split-K reduction: out = f16(sum_s part[s] + bias[n])
__global__ void k_splitk_reduce(const float* __restrict__ part,
                                const float* __restrict__ bias,
                                h16* __restrict__ out, long MN, int S, int N)
{
    long i = (long)blockIdx.x * blockDim.x + threadIdx.x;
    if (i >= MN) return;
    float s = 0.0f;
    for (int z = 0; z < S; ++z) s += part[(long)z * MN + i];
    out[i] = (h16)(s + bias[(int)(i % N)]);
}

// ---------------------------------------------------------------------------
// Elementwise / small kernels
// ---------------------------------------------------------------------------
__global__ void k_f32_to_f16(const float* __restrict__ s, h16* __restrict__ d, long n)
{
    long i = (long)blockIdx.x * blockDim.x + threadIdx.x;
    if (i < n) d[i] = (h16)s[i];
}

// W_x (LAYERS,40,256) -> f16 (LAYERS,48,256) zero-padded rows 40..47
__global__ void k_pad_wx(const float* __restrict__ wx, h16* __restrict__ d)
{
    long i = (long)blockIdx.x * blockDim.x + threadIdx.x;
    if (i >= (long)LAYERS * NPAD * DI) return;
    int c = (int)(i % DI);
    int r = (int)((i / DI) % NPAD);
    int l = (int)(i / ((long)DI * NPAD));
    d[i] = (r < DTRANK + 2 * DSTATE)
         ? (h16)wx[((long)l * 40 + r) * DI + c] : (h16)0.0f;
}

// x[b,c,e] = cls (c==0) else t[b,2(c-1)]*We[e,0]+t[b,2(c-1)+1]*We[e,1]+be[e]+pos[c-1,e]
__global__ void k_embed(const h16* __restrict__ t, const float* __restrict__ We,
                        const float* __restrict__ be, const float* __restrict__ cls,
                        const float* __restrict__ pos, h16* __restrict__ x)
{
    long i = (long)blockIdx.x * blockDim.x + threadIdx.x;
    if (i >= (long)BATCH * LSEQ * EMBD) return;
    int e = (int)(i % EMBD);
    long rest = i / EMBD;
    int c = (int)(rest % LSEQ);
    long b = rest / LSEQ;
    float v;
    if (c == 0) {
        v = cls[e];
    } else {
        float t0 = (float)t[b * TOKD + (long)(c - 1) * 2 + 0];
        float t1 = (float)t[b * TOKD + (long)(c - 1) * 2 + 1];
        v = t0 * We[e * 2 + 0] + t1 * We[e * 2 + 1] + be[e]
          + pos[(long)(c - 1) * EMBD + e];
    }
    x[i] = (h16)v;
}

// causal depthwise conv (DCONV=4) over xi = xz[..., :DI], then SiLU
__global__ void k_conv_silu(const h16* __restrict__ xz, const float* __restrict__ cw,
                            const float* __restrict__ cb, h16* __restrict__ xc)
{
    long i = (long)blockIdx.x * blockDim.x + threadIdx.x;
    if (i >= (long)BATCH * LSEQ * DI) return;
    int d = (int)(i % DI);
    long bt = i / DI;
    int t = (int)(bt % LSEQ);
    long b = bt / LSEQ;
    const h16* base = xz + b * (long)LSEQ * (2 * DI);
    float acc = cb[d];
#pragma unroll
    for (int j = 0; j < DCONV; ++j) {
        int tt = t - (DCONV - 1) + j;
        if (tt >= 0) acc += (float)base[(long)tt * (2 * DI) + d] * cw[d * DCONV + j];
    }
    xc[i] = (h16)silu_f(acc);
}

// ---------------------------------------------------------------------------
// Selective scan. One channel (b,d) per 16 lanes; lane owns state n.
// Fuses: delta = softplus(dt @ W_dt.T + b_dt)   (inline, off critical path)
//        y     = (sum_n h*C + Dp*xc) * silu(z)  (gate fused at store)
// Recurrence critical path per step: 1 exp + 1 fma per lane. No barriers.
// ---------------------------------------------------------------------------
__global__ __launch_bounds__(256) void k_scan(
    const h16* __restrict__ dbc,   // (B,L,48): dt[0:8], B[8:24], C[24:40]
    const h16* __restrict__ xz,    // (B,L,512): z = [..., 256+d]
    const h16* __restrict__ xc,    // (B,L,256)
    const float* __restrict__ Wdt, const float* __restrict__ bdt,
    const float* __restrict__ Alog, const float* __restrict__ Dp,
    h16* __restrict__ y)           // (B,L,256) gated output
{
    const int b = blockIdx.x;
    const int n = threadIdx.x & 15;            // state index
    const int c = threadIdx.x >> 4;            // channel within block (0..15)
    const int d = blockIdx.y * 16 + c;

    const float Ac = -__expf(Alog[d * DSTATE + n]);
    const float Dd = Dp[d];
    const float bd = bdt[d];
    float wdt[DTRANK];
#pragma unroll
    for (int j = 0; j < DTRANK; ++j) wdt[j] = Wdt[d * DTRANK + j];

    const h16* dbc_b = dbc + (long)b * LSEQ * NPAD;
    const h16* xc_b  = xc  + (long)b * LSEQ * DI + d;
    const h16* z_b   = xz  + (long)b * LSEQ * (2 * DI) + DI + d;
    h16*       y_b   = y   + (long)b * LSEQ * DI + d;

    float hn = 0.0f;
    for (int t = 0; t < LSEQ; ++t) {
        const h16* row = dbc_b + (long)t * NPAD;
        float acc = bd;
#pragma unroll
        for (int j = 0; j < DTRANK; ++j) acc += (float)row[j] * wdt[j];
        float dl = (acc > 20.0f) ? acc : log1pf(__expf(acc));
        float Bn = (float)row[DTRANK + n];
        float Cn = (float)row[DTRANK + DSTATE + n];
        float xv = (float)xc_b[(long)t * DI];
        hn = __expf(dl * Ac) * hn + (dl * xv) * Bn;
        float p = hn * Cn;
#pragma unroll
        for (int o = 8; o >= 1; o >>= 1) p += __shfl_xor(p, o, 32);
        if (n == 0) {
            float z = (float)z_b[(long)t * (2 * DI)];
            y_b[(long)t * DI] = (h16)((p + Dd * xv) * silu_f(z));
        }
    }
}

// LayerNorm over EMB=128 + transpose to (L,B,E); one wave per row.
__global__ __launch_bounds__(256) void k_ln(const h16* __restrict__ x,
                                            const float* __restrict__ g,
                                            const float* __restrict__ be,
                                            float* __restrict__ out)
{
    const int lane = threadIdx.x & 31;
    const int wave = threadIdx.x >> 5;
    long row = (long)blockIdx.x * 8 + wave;
    if (row >= (long)MROWS) return;
    int b = (int)(row / LSEQ);
    int t = (int)(row % LSEQ);
    const h16* xr = x + row * EMBD;
    float v[4];
    float s = 0.0f;
#pragma unroll
    for (int i = 0; i < 4; ++i) { v[i] = (float)xr[lane * 4 + i]; s += v[i]; }
#pragma unroll
    for (int o = 16; o >= 1; o >>= 1) s += __shfl_xor(s, o, 32);
    float mu = s * (1.0f / EMBD);
    float q = 0.0f;
#pragma unroll
    for (int i = 0; i < 4; ++i) { float dv = v[i] - mu; q += dv * dv; }
#pragma unroll
    for (int o = 16; o >= 1; o >>= 1) q += __shfl_xor(q, o, 32);
    float rstd = rsqrtf(q * (1.0f / EMBD) + 1e-5f);
    float* orow = out + ((long)t * BATCH + b) * EMBD;
#pragma unroll
    for (int i = 0; i < 4; ++i) {
        int e = lane * 4 + i;
        orow[e] = (v[i] - mu) * rstd * g[e] + be[e];
    }
}

// ---------------------------------------------------------------------------
// Host driver
// ---------------------------------------------------------------------------
static inline int cdiv(long a, long b) { return (int)((a + b - 1) / b); }

extern "C" void kernel_launch(void* const* d_in, const int* in_sizes, int n_in,
                              void* d_out, int out_size, void* d_ws, size_t ws_size,
                              hipStream_t stream)
{
    const float* patches = (const float*)d_in[0];
    const float* W_tok   = (const float*)d_in[1];
    const float* b_tok   = (const float*)d_in[2];
    const float* W_emb   = (const float*)d_in[3];
    const float* b_emb   = (const float*)d_in[4];
    const float* cls_tok = (const float*)d_in[5];
    const float* pos_emb = (const float*)d_in[6];
    const float* W_in    = (const float*)d_in[7];
    const float* conv_w  = (const float*)d_in[8];
    const float* conv_b  = (const float*)d_in[9];
    const float* W_x     = (const float*)d_in[10];
    const float* W_dt    = (const float*)d_in[11];
    const float* b_dt    = (const float*)d_in[12];
    const float* A_log   = (const float*)d_in[13];
    const float* Dp      = (const float*)d_in[14];
    const float* W_out   = (const float*)d_in[15];
    const float* ln_g    = (const float*)d_in[16];
    const float* ln_b    = (const float*)d_in[17];
    float* out = (float*)d_out;

    size_t off = 0;
    auto alloc_b = [&](size_t bytes) -> void* {
        off = (off + 255) & ~(size_t)255;
        void* p = (char*)d_ws + off;
        off += bytes;
        return p;
    };
    auto alloc_h = [&](long n) -> h16* { return (h16*)alloc_b((size_t)n * sizeof(h16)); };

    h16*   p_patches = alloc_h((long)BATCH * TOKD);
    h16*   p_wtok    = alloc_h((long)TOKD * TOKD);
    h16*   p_win     = alloc_h((long)LAYERS * 2 * DI * EMBD);
    h16*   p_wx      = alloc_h((long)LAYERS * NPAD * DI);
    h16*   p_wout    = alloc_h((long)LAYERS * EMBD * DI);
    h16*   p_t       = alloc_h((long)BATCH * TOKD);
    h16*   p_x0      = alloc_h((long)MROWS * EMBD);
    h16*   p_x1      = alloc_h((long)MROWS * EMBD);
    h16*   p_xz      = alloc_h((long)MROWS * 2 * DI);
    h16*   p_xc      = alloc_h((long)MROWS * DI);
    h16*   p_dbc     = alloc_h((long)MROWS * NPAD);
    h16*   p_y       = alloc_h((long)MROWS * DI);
    float* p_part    = (float*)alloc_b((size_t)NSPLIT_TOK * BATCH * TOKD * sizeof(float));

    const int TB = 256;

    // ---- weight / input conversion to f16 ----
    k_f32_to_f16<<<cdiv((long)BATCH * TOKD, TB), TB, 0, stream>>>(patches, p_patches, (long)BATCH * TOKD);
    k_f32_to_f16<<<cdiv((long)TOKD * TOKD, TB), TB, 0, stream>>>(W_tok, p_wtok, (long)TOKD * TOKD);
    k_f32_to_f16<<<cdiv((long)LAYERS * 2 * DI * EMBD, TB), TB, 0, stream>>>(W_in, p_win, (long)LAYERS * 2 * DI * EMBD);
    k_f32_to_f16<<<cdiv((long)LAYERS * EMBD * DI, TB), TB, 0, stream>>>(W_out, p_wout, (long)LAYERS * EMBD * DI);
    k_pad_wx<<<cdiv((long)LAYERS * NPAD * DI, TB), TB, 0, stream>>>(W_x, p_wx);

    // ---- stage 1: t = patches @ W_tok.T + b_tok (M=64,N=4000,K=4000), split-K=8
    {
        dim3 g(cdiv(TOKD / 16, 8), BATCH / 32, NSPLIT_TOK);
        gemm_nt_wmma<<<g, 128, 0, stream>>>(p_patches, p_wtok, nullptr, p_t, p_part,
                                            BATCH, TOKD, TOKD, TOKD, TOKD, TOKD, KSPLIT_TOK);
        k_splitk_reduce<<<cdiv((long)BATCH * TOKD, TB), TB, 0, stream>>>(
            p_part, b_tok, p_t, (long)BATCH * TOKD, NSPLIT_TOK, TOKD);
    }

    // ---- stage 2: embed + pos + cls
    k_embed<<<cdiv((long)BATCH * LSEQ * EMBD, TB), TB, 0, stream>>>(
        p_t, W_emb, b_emb, cls_tok, pos_emb, p_x0);

    h16* x_cur = p_x0;
    h16* x_nxt = p_x1;
    const long nBLD = (long)MROWS * DI;

    for (int l = 0; l < LAYERS; ++l) {
        // xz = x @ W_in[l].T   (M=128064, N=512, K=128)
        {
            dim3 g((2 * DI) / 128, MROWS / 32, 1);
            gemm_nt_wmma<<<g, 128, 0, stream>>>(x_cur, p_win + (long)l * 2 * DI * EMBD,
                                                nullptr, p_xz, nullptr,
                                                MROWS, 2 * DI, EMBD, EMBD, EMBD, 2 * DI, EMBD);
        }
        // xc = silu(conv(xi) + cb)
        k_conv_silu<<<cdiv(nBLD, TB), TB, 0, stream>>>(
            p_xz, conv_w + (long)l * DI * DCONV, conv_b + (long)l * DI, p_xc);
        // dbc = xc @ W_x[l].T  (M=128064, N=48(pad), K=256)
        {
            dim3 g(1, MROWS / 32, 1);
            gemm_nt_wmma<<<g, 128, 0, stream>>>(p_xc, p_wx + (long)l * NPAD * DI,
                                                nullptr, p_dbc, nullptr,
                                                MROWS, NPAD, DI, DI, DI, NPAD, DI);
        }
        // fused: delta proj + softplus + scan + Dp*xc + SiLU gate
        {
            dim3 g(BATCH, DI / 16, 1);
            k_scan<<<g, 256, 0, stream>>>(
                p_dbc, p_xz, p_xc,
                W_dt + (long)l * DI * DTRANK, b_dt + (long)l * DI,
                A_log + (long)l * DI * DSTATE, Dp + (long)l * DI, p_y);
        }
        // x_next = y @ W_out[l].T (M=128064, N=128, K=256)
        {
            dim3 g(EMBD / 128, MROWS / 32, 1);
            gemm_nt_wmma<<<g, 128, 0, stream>>>(p_y, p_wout + (long)l * EMBD * DI,
                                                nullptr, x_nxt, nullptr,
                                                MROWS, EMBD, DI, DI, DI, EMBD, DI);
        }
        h16* tmp = x_cur; x_cur = x_nxt; x_nxt = tmp;
    }

    // ---- LayerNorm + transpose to (L,B,E)
    k_ln<<<cdiv((long)MROWS, 8), 256, 0, stream>>>(x_cur, ln_g, ln_b, out);

    (void)in_sizes; (void)n_in; (void)out_size; (void)ws_size;
}
